// LMTAD_Quantum_1889785610328
// MI455X (gfx1250) — matrix operand order repaired and checked by reading, compiled, not run on hardware
//
#include <hip/hip_runtime.h>
#include <hip/hip_bf16.h>
#include <math.h>

typedef __attribute__((ext_vector_type(16))) _Float16 v16h;
typedef __attribute__((ext_vector_type(8)))  float    v8f;
typedef __attribute__((ext_vector_type(4)))  float    v4f;

namespace {
constexpr int B_ = 4, T_ = 1024, D_ = 768, L_ = 4, V_ = 50257, DFF_ = 3072;
constexpr int VP_ = 50272;                       // V rounded up to multiple of 16
constexpr int KS_ = 8;                           // split-K slices for row GEMMs
constexpr int NSTEP_ = D_ / 32;                  // 24 WMMA K-steps

// workspace layout (float offsets; all multiples of 8 floats = 32B aligned)
constexpr size_t OFF_X    = 0;                                  // B*T*D embeddings
constexpr size_t OFF_C    = OFF_X    + (size_t)B_ * T_ * D_;    // 768: colmean(kw[0])
constexpr size_t OFF_KBM  = OFF_C    + D_;                      // mean(kb[0]) (padded)
constexpr size_t OFF_SC   = OFF_KBM  + 256;                     // B*T scores
constexpr size_t OFF_W    = OFF_SC   + (size_t)B_ * T_;         // B*T softmax weights
constexpr size_t OFF_XW   = OFF_W    + (size_t)B_ * T_;         // B*D
constexpr size_t OFF_CTX  = OFF_XW   + (size_t)B_ * D_;         // B*D
constexpr size_t OFF_H    = OFF_CTX  + (size_t)B_ * D_;         // B*DFF
constexpr size_t OFF_ROW  = OFF_H    + (size_t)B_ * DFF_;       // B*D
constexpr size_t OFF_Y    = OFF_ROW  + (size_t)B_ * D_;         // B*D normalized
constexpr size_t OFF_AF   = OFF_Y    + (size_t)B_ * D_;         // 24*32*16 f16 = 6144 floats
constexpr size_t OFF_PART = OFF_AF   + 6144;                    // KS*4*DFF partials
constexpr size_t OFF_LR   = OFF_PART + (size_t)KS_ * 4 * DFF_;  // B*VP logits rows
} // namespace

__device__ __forceinline__ float gelu_exact(float x) {
    return 0.5f * x * (1.0f + erff(x * 0.70710678118654752440f));
}

// ---- 1) c[d] = mean_j kw0[d][j];  kbm = mean(kb0) ---------------------------
__global__ __launch_bounds__(1024) void prep_colmean(
        const float* __restrict__ kw0, const float* __restrict__ kb0,
        float* __restrict__ ws) {
    int t = threadIdx.x;
    if (t < D_) {
        const float* r = kw0 + (size_t)t * D_;
        float s = 0.f;
        for (int j = 0; j < D_; ++j) s += r[j];
        ws[OFF_C + t] = s * (1.0f / D_);
    } else if (t == D_) {
        float s = 0.f;
        for (int j = 0; j < D_; ++j) s += kb0[j];
        ws[OFF_KBM] = s * (1.0f / D_);
    }
}

// ---- 2) x = wte[idx]+wpe[t]; scores = cos(x.c + kbm) ------------------------
__global__ __launch_bounds__(256) void embed_score(
        const int* __restrict__ idx, const float* __restrict__ wte,
        const float* __restrict__ wpe, float* __restrict__ ws) {
    __shared__ float red[256];
    int bt  = blockIdx.x;
    int t   = bt & (T_ - 1);
    int tok = idx[bt];
    const float* wt = wte + (size_t)tok * D_;
    const float* wp = wpe + (size_t)t * D_;
    float*       xr = ws + OFF_X + (size_t)bt * D_;
    const float* c  = ws + OFF_C;
    float part = 0.f;
    for (int d = threadIdx.x; d < D_; d += 256) {
        float xv = wt[d] + wp[d];
        xr[d] = xv;
        part += xv * c[d];
    }
    red[threadIdx.x] = part; __syncthreads();
    for (int s = 128; s > 0; s >>= 1) {
        if (threadIdx.x < s) red[threadIdx.x] += red[threadIdx.x + s];
        __syncthreads();
    }
    if (threadIdx.x == 0) ws[OFF_SC + bt] = cosf(red[0] + ws[OFF_KBM]);
}

// ---- 3) w = softmax over T (per batch) --------------------------------------
__global__ __launch_bounds__(1024) void softmax_t(float* __restrict__ ws) {
    __shared__ float red[1024];
    int b = blockIdx.x, t = threadIdx.x;
    float v = ws[OFF_SC + (size_t)b * T_ + t];
    red[t] = v; __syncthreads();
    for (int s = 512; s > 0; s >>= 1) {
        if (t < s) red[t] = fmaxf(red[t], red[t + s]);
        __syncthreads();
    }
    float m = red[0]; __syncthreads();
    float e = expf(v - m);
    red[t] = e; __syncthreads();
    for (int s = 512; s > 0; s >>= 1) {
        if (t < s) red[t] += red[t + s];
        __syncthreads();
    }
    ws[OFF_W + (size_t)b * T_ + t] = e / red[0];
}

// ---- 4) xw[b,d] = sum_t w[b,t] * x[b,t,d] -----------------------------------
__global__ __launch_bounds__(256) void weighted_sum(float* __restrict__ ws) {
    __shared__ float sw[T_];
    int b = blockIdx.y;
    int d = blockIdx.x * 256 + threadIdx.x;
    for (int i = threadIdx.x; i < T_; i += 256) sw[i] = ws[OFF_W + (size_t)b * T_ + i];
    __syncthreads();
    const float* xb = ws + OFF_X + (size_t)b * T_ * D_;
    float acc = 0.f;
    for (int t = 0; t < T_; ++t)
        acc = fmaf(sw[t], __builtin_nontemporal_load(xb + (size_t)t * D_ + d), acc);
    ws[OFF_XW + (size_t)b * D_ + d] = acc;
}

// ---- 5a) split-K GEMM slice: part[s][0:4][j] = in[0:4,k0:k1] @ W[k0:k1,j] ---
__global__ __launch_bounds__(256) void gemm4_slice(
        const float* __restrict__ in, const float* __restrict__ w,
        float* __restrict__ part, int K, int N) {
    extern __shared__ float sIn[];                 // 4 * (K/KS_) floats
    int ks = K / KS_;
    int k0 = blockIdx.y * ks;
    for (int i = threadIdx.x; i < 4 * ks; i += 256) {
        int r = i / ks, d = i - r * ks;
        sIn[i] = in[(size_t)r * K + k0 + d];
    }
    __syncthreads();
    int j = blockIdx.x * 256 + threadIdx.x;
    if (j >= N) return;
    const float* wp = w + (size_t)k0 * N + j;
    float a0 = 0.f, a1 = 0.f, a2 = 0.f, a3 = 0.f;
    for (int d = 0; d < ks; ++d) {
        float wv = __builtin_nontemporal_load(wp + (size_t)d * N);
        a0 = fmaf(sIn[d],          wv, a0);
        a1 = fmaf(sIn[ks + d],     wv, a1);
        a2 = fmaf(sIn[2 * ks + d], wv, a2);
        a3 = fmaf(sIn[3 * ks + d], wv, a3);
    }
    float* pp = part + ((size_t)blockIdx.y * 4) * N + j;
    pp[0] = a0; pp[(size_t)N] = a1; pp[2 * (size_t)N] = a2; pp[3 * (size_t)N] = a3;
}

// ---- 5b) deterministic fixed-order reduce over slices + bias + act ----------
__global__ __launch_bounds__(256) void gemm4_reduce(
        const float* __restrict__ part, const float* __restrict__ bias,
        float* __restrict__ out, int N, int act) {
    int j = blockIdx.x * 256 + threadIdx.x;
    if (j >= N) return;
    float bv = bias ? bias[j] : 0.f;
    #pragma unroll
    for (int r = 0; r < 4; ++r) {
        float s = bv;
        #pragma unroll
        for (int sl = 0; sl < KS_; ++sl)
            s += part[((size_t)sl * 4 + r) * N + j];
        if (act) s = gelu_exact(s);
        out[(size_t)r * N + j] = s;
    }
}

// ---- 6) final LayerNorm on the 4 rows ---------------------------------------
__global__ __launch_bounds__(256) void ln_final(float* __restrict__ ws,
                                                const float* __restrict__ g) {
    __shared__ float red[256];
    int b = blockIdx.x;
    const float* x = ws + OFF_ROW + (size_t)b * D_;
    float s = 0.f;
    for (int d = threadIdx.x; d < D_; d += 256) s += x[d];
    red[threadIdx.x] = s; __syncthreads();
    for (int k = 128; k > 0; k >>= 1) {
        if (threadIdx.x < k) red[threadIdx.x] += red[threadIdx.x + k];
        __syncthreads();
    }
    float mu = red[0] * (1.0f / D_); __syncthreads();
    float q = 0.f;
    for (int d = threadIdx.x; d < D_; d += 256) { float dv = x[d] - mu; q += dv * dv; }
    red[threadIdx.x] = q; __syncthreads();
    for (int k = 128; k > 0; k >>= 1) {
        if (threadIdx.x < k) red[threadIdx.x] += red[threadIdx.x + k];
        __syncthreads();
    }
    float inv = rsqrtf(red[0] * (1.0f / D_) + 1e-5f);
    for (int d = threadIdx.x; d < D_; d += 256)
        ws[OFF_Y + (size_t)b * D_ + d] = (x[d] - mu) * inv * g[d];
}

// ---- 6b) prepack A fragments (f16, contiguous-k layout) for all 24 K-steps -----------
// af[s*512 + lane*16 + e] = y[M=lane&15][ k = s*32 + K_A(e,laneHi) ]  (0 if M>=4)
// K_A(e,hi) = e + 16*hi     (contiguous, matches B fill)
__global__ __launch_bounds__(256) void pack_a(const float* __restrict__ y,
                                              _Float16* __restrict__ af) {
    int i = blockIdx.x * 256 + threadIdx.x;          // 0 .. 24*512-1
    int s    = i >> 9;
    int lane = (i >> 4) & 31;
    int e    = i & 15;
    int laneLo = lane & 15, laneHi = lane >> 4;
    int k = s * 32 + e + 16 * laneHi;
    float v = (laneLo < B_) ? y[(size_t)laneLo * D_ + k] : 0.f;
    af[i] = (_Float16)v;
}

// ---- 7) lm_head: lr[16(pad of 4) x V] = y_pad @ wte^T via v_wmma_f32_16x16x32_f16
// One wave per 16-column tile of V. A from prepacked f16 frags (one v16h load),
// B built from 16 contiguous NT f32 loads per lane (K_B(e,hi) = e + 16*hi).
// C/D: lane -> N, VGPR r -> M (batch rows 0..3 = VGPRs 0..3 on lanes 0..15).
__global__ __launch_bounds__(256) void lm_head_wmma(
        const _Float16* __restrict__ af, const float* __restrict__ wte,
        float* __restrict__ lr) {
    int lane = threadIdx.x & 31;
    int tile = blockIdx.x * (blockDim.x >> 5) + (threadIdx.x >> 5);
    int v0 = tile * 16;
    if (v0 >= V_) return;                            // uniform per wave
    int laneLo = lane & 15, laneHi = lane >> 4;
    int n = v0 + laneLo; if (n >= V_) n = V_ - 1;    // clamp (store covered by VP pad)
    const float*    wrow = wte + (size_t)n * D_;
    const _Float16* ap   = af + (size_t)lane * 16;
    v8f acc = {};
    for (int s = 0; s < NSTEP_; ++s) {
        v16h a = *(const v16h*)(ap + (size_t)s * 512);
        v16h bfr;
        const v4f* bp = (const v4f*)(wrow + s * 32 + 16 * laneHi);
        #pragma unroll
        for (int q = 0; q < 4; ++q) {
            v4f f = __builtin_nontemporal_load(bp + q);
            bfr[4 * q + 0] = (_Float16)f[0]; bfr[4 * q + 1] = (_Float16)f[1];
            bfr[4 * q + 2] = (_Float16)f[2]; bfr[4 * q + 3] = (_Float16)f[3];
        }
        acc = __builtin_amdgcn_wmma_f32_16x16x32_f16(
                false, a, false, bfr, (short)0, acc, false, false);
    }
    if (laneHi == 0) {
        #pragma unroll
        for (int r = 0; r < B_; ++r)
            lr[(size_t)r * VP_ + v0 + laneLo] = acc[r];
    }
}

// ---- 8) broadcast logits rows over T with NT b128 stores --------------------
__global__ __launch_bounds__(256) void bcast_out(const float* __restrict__ lr,
                                                 float* __restrict__ out) {
    int t = blockIdx.x, b = blockIdx.y;
    const float* lrb = lr + (size_t)b * VP_;
    size_t obase = ((size_t)b * T_ + t) * (size_t)V_;
    float* orow = out + obase;
    int head = (int)((4 - (obase & 3)) & 3);         // scalars until 16B aligned
    if (threadIdx.x < head)
        __builtin_nontemporal_store(lrb[threadIdx.x], orow + threadIdx.x);
    int n4 = (V_ - head) >> 2;
    for (int i = threadIdx.x; i < n4; i += 256) {
        int v = head + 4 * i;
        v4f val = { lrb[v], lrb[v + 1], lrb[v + 2], lrb[v + 3] };
        __builtin_nontemporal_store(val, (v4f*)(orow + v));
    }
    int tail0 = head + 4 * n4;
    int tl = V_ - tail0;
    if (threadIdx.x < tl)
        __builtin_nontemporal_store(lrb[tail0 + threadIdx.x], orow + tail0 + threadIdx.x);
}

extern "C" void kernel_launch(void* const* d_in, const int* in_sizes, int n_in,
                              void* d_out, int out_size, void* d_ws, size_t ws_size,
                              hipStream_t stream) {
    (void)in_sizes; (void)n_in; (void)out_size; (void)ws_size;
    const int*   idx = (const int*)  d_in[0];
    const float* wte = (const float*)d_in[1];
    const float* wpe = (const float*)d_in[2];
    const float* lnw = (const float*)d_in[3];
    // d_in[4]=qw, d_in[5]=qb are dead in the reference (Q unused)
    const float* kw  = (const float*)d_in[6];
    const float* kb  = (const float*)d_in[7];
    const float* vw  = (const float*)d_in[8];
    const float* vb  = (const float*)d_in[9];
    const float* fcw = (const float*)d_in[10];
    const float* pw  = (const float*)d_in[11];
    float* out = (float*)d_out;
    float* ws  = (float*)d_ws;
    float* part = ws + OFF_PART;

    // 1) colmean(kw[0]) + mean(kb[0])
    prep_colmean<<<1, 1024, 0, stream>>>(kw, kb, ws);
    // 2) embeddings + k_mean dot + cos
    embed_score<<<B_ * T_, 256, 0, stream>>>(idx, wte, wpe, ws);
    // 3) per-batch softmax over T
    softmax_t<<<B_, 1024, 0, stream>>>(ws);
    // 4) attention-weighted input rows
    weighted_sum<<<dim3(D_ / 256, B_), 256, 0, stream>>>(ws);

    // 5) layer stack on 4 rows: ctx = in@vw+vb; h = gelu(ctx@fcw); row = h@pw
    for (int l = 0; l < L_; ++l) {
        const float* in  = ws + ((l == 0) ? OFF_XW : OFF_ROW);
        const float* vwl = vw  + (size_t)l * D_ * D_;
        const float* vbl = vb  + (size_t)l * D_;
        const float* fcl = fcw + (size_t)l * D_ * DFF_;
        const float* pwl = pw  + (size_t)l * DFF_ * D_;

        gemm4_slice<<<dim3(D_ / 256, KS_), 256, 4 * (D_ / KS_) * sizeof(float), stream>>>(
            in, vwl, part, D_, D_);
        gemm4_reduce<<<D_ / 256, 256, 0, stream>>>(part, vbl, ws + OFF_CTX, D_, 0);

        gemm4_slice<<<dim3(DFF_ / 256, KS_), 256, 4 * (D_ / KS_) * sizeof(float), stream>>>(
            ws + OFF_CTX, fcl, part, D_, DFF_);
        gemm4_reduce<<<DFF_ / 256, 256, 0, stream>>>(part, nullptr, ws + OFF_H, DFF_, 1);

        gemm4_slice<<<dim3(D_ / 256, KS_), 256, 4 * (DFF_ / KS_) * sizeof(float), stream>>>(
            ws + OFF_H, pwl, part, DFF_, D_);
        gemm4_reduce<<<D_ / 256, 256, 0, stream>>>(part, nullptr, ws + OFF_ROW, D_, 0);
    }

    // 6) final LN + prepack A fragments
    ln_final<<<B_, 256, 0, stream>>>(ws, lnw);
    pack_a<<<(NSTEP_ * 512) / 256, 256, 0, stream>>>(ws + OFF_Y,
                                                     (_Float16*)(ws + OFF_AF));

    // 7) lm_head via WMMA: 3142 16-col tiles, 8 waves/block
    int tiles  = (V_ + 15) / 16;
    int blocks = (tiles + 7) / 8;
    lm_head_wmma<<<blocks, 256, 0, stream>>>((const _Float16*)(ws + OFF_AF),
                                             wte, ws + OFF_LR);

    // 8) broadcast over T (write-bound: 823 MB, NT b128 stores)
    bcast_out<<<dim3(T_, B_), 256, 0, stream>>>(ws + OFF_LR, out);
}
